// SelfSupervisedMamba2_63720134803770
// MI455X (gfx1250) — compile-verified, matrix operand order checked by reading, not run on hardware
//
#include <hip/hip_runtime.h>
#include <hip/hip_bf16.h>

// ---------------------------------------------------------------------------
// Problem constants (from the reference)
// ---------------------------------------------------------------------------
#define D_MODEL   256
#define D_STATE   64
#define D_CONV    4
#define HEADDIM   64
#define D_INNER   512              // EXPAND * D_MODEL
#define NHEADS    8                // D_INNER / HEADDIM
#define CONV_DIM  640              // D_INNER + 2*D_STATE
#define D_IN_PROJ 1160             // 2*D_INNER + 2*D_STATE + NHEADS
#define BATCH     4
#define SEQLEN    2048
#define NTOK      (BATCH * SEQLEN) // 8192
#define EPS       1e-5f
#define N_INPROJ_PAD 1216          // 1160 rounded up to multiple of 64

typedef __attribute__((ext_vector_type(16))) __bf16 v16bf;
typedef __attribute__((ext_vector_type(8)))  __bf16 v8bf;
typedef __attribute__((ext_vector_type(8)))  float  v8f;
typedef __attribute__((ext_vector_type(4)))  int    v4i;

// ---------------------------------------------------------------------------
// Async global->LDS copy (CDNA5 GLOBAL_LOAD_ASYNC_TO_LDS_B128, ASYNCcnt),
// with a synchronous fallback if the toolchain lacks the builtin.
// Probe-derived signature: (v4i __device__* gsrc, v4i __shared__* ldst,
//                           imm int offset, imm int cpol)
// ---------------------------------------------------------------------------
#if defined(__HIP_DEVICE_COMPILE__) && defined(__gfx1250__) && \
    __has_builtin(__builtin_amdgcn_global_load_async_to_lds_b128)
#define HAVE_ASYNC_LDS 1
#endif

__device__ __forceinline__ void lds_copy_b128(const __bf16* g, __bf16* l) {
#ifdef HAVE_ASYNC_LDS
    // AS1 pointer: same 64-bit value as the generic (flat) global pointer.
    union { const __bf16* p; __attribute__((address_space(1))) v4i* g; } gc;
    gc.p = g;
    // AS3 pointer: low 32 bits of the flat LDS-aperture address are the LDS offset.
    union { unsigned long long u; __attribute__((address_space(3))) v4i* l; } lc;
    lc.u = (unsigned long long)(uintptr_t)l;
    __builtin_amdgcn_global_load_async_to_lds_b128(gc.g, lc.l, 0, 0);
#else
    *(v8bf*)l = *(const v8bf*)g;
#endif
}

__device__ __forceinline__ void lds_copy_wait() {
#ifdef HAVE_ASYNC_LDS
#if __has_builtin(__builtin_amdgcn_s_wait_asynccnt)
    __builtin_amdgcn_s_wait_asynccnt(0);
#else
    asm volatile("s_wait_asynccnt 0x0" ::: "memory");
#endif
#endif
}

// ---------------------------------------------------------------------------
// f32 -> bf16 weight conversion with zero row padding (rows padded to padRows)
// ---------------------------------------------------------------------------
__global__ void cvt_weight_kernel(const float* __restrict__ in, __bf16* __restrict__ out,
                                  int rows, int K, int padRows) {
    int idx = blockIdx.x * blockDim.x + threadIdx.x;
    int total = padRows * K;
    if (idx >= total) return;
    int r = idx / K;
    out[idx] = (r < rows) ? (__bf16)in[idx] : (__bf16)0.0f;
}

// f32 -> bf16 activation conversion, optional leaky-relu(0.01)
__global__ void cvt_act_kernel(const float* __restrict__ in, __bf16* __restrict__ out,
                               int n, int leaky) {
    int idx = blockIdx.x * blockDim.x + threadIdx.x;
    if (idx >= n) return;
    float v = in[idx];
    if (leaky) v = (v > 0.0f) ? v : 0.01f * v;
    out[idx] = (__bf16)v;
}

// ---------------------------------------------------------------------------
// WMMA bf16 GEMM:  C[M,N] (f32) = A[M,K] (bf16, row-major) * W[N,K]^T (bf16)
//
// Block = 8 waves (wave32), block tile 256(M) x 64(N); wave tile 32x64
// (2 A fragments x 4 B tiles = 8 WMMAs per 32-wide k-step).
// The 64-row B panel is shared by all 8 waves, so it is staged in LDS once
// per block (K-chunks of 256) via async global->LDS copies; inner-loop B
// fragment reads are conflict-free ds_load_b128 (row pitch 264 bf16).
// Requires: M % 256 == 0, K % 256 == 0, W row-padded to gridDim.x*64 rows.
// ---------------------------------------------------------------------------
#define KCHUNK    256
#define LDS_PITCH (KCHUNK + 8)

__global__ void __launch_bounds__(256)
gemm_wmma_kernel(const __bf16* __restrict__ A, const __bf16* __restrict__ W,
                 float* __restrict__ C, const float* __restrict__ bias,
                 int M, int N, int K) {
    const int lane  = threadIdx.x;            // 0..31 (wave32)
    const int wave  = threadIdx.y;            // 0..7
    const int tid   = wave * 32 + lane;
    const int mBase = blockIdx.y * 256 + wave * 32;
    const int nBase = blockIdx.x * 64;

    // CDNA5 16-bit A/B fragment layout: lane l<16 -> row l, K {ks..ks+7, 16+ks..+7}
    const int r  = lane & 15;
    const int ks = (lane >> 4) * 8;

    __shared__ __bf16 sB[64][LDS_PITCH];

    const __bf16* aRow0 = A + (size_t)(mBase + r) * K + ks;
    const __bf16* aRow1 = aRow0 + (size_t)16 * K;

    v8f acc[2][4];
    #pragma unroll
    for (int mf = 0; mf < 2; ++mf)
        #pragma unroll
        for (int t = 0; t < 4; ++t)
            #pragma unroll
            for (int i = 0; i < 8; ++i) acc[mf][t][i] = 0.0f;

    for (int kb = 0; kb < K; kb += KCHUNK) {
        // Stage B panel chunk: 64 rows x KCHUNK cols, 16B per thread per step.
        #pragma unroll
        for (int j = tid; j < 64 * (KCHUNK / 8); j += 256) {
            int row = j >> 5;           // KCHUNK/8 == 32 chunks per row
            int cc  = (j & 31) << 3;
            lds_copy_b128(W + (size_t)(nBase + row) * K + kb + cc, &sB[row][cc]);
        }
        lds_copy_wait();
        __syncthreads();

        for (int ko = 0; ko < KCHUNK; ko += 32) {
            union { v16bf v; v8bf h[2]; } a0, a1;
            const __bf16* ap0 = aRow0 + kb + ko;
            const __bf16* ap1 = aRow1 + kb + ko;
            a0.h[0] = *(const v8bf*)(ap0);
            a0.h[1] = *(const v8bf*)(ap0 + 16);
            a1.h[0] = *(const v8bf*)(ap1);
            a1.h[1] = *(const v8bf*)(ap1 + 16);
            #pragma unroll
            for (int t = 0; t < 4; ++t) {
                union { v16bf v; v8bf h[2]; } b;
                const __bf16* bp = &sB[t * 16 + r][ko + ks];
                b.h[0] = *(const v8bf*)(bp);
                b.h[1] = *(const v8bf*)(bp + 16);
                acc[0][t] = __builtin_amdgcn_wmma_f32_16x16x32_bf16(
                    false, a0.v, false, b.v, (short)0, acc[0][t], false, false);
                acc[1][t] = __builtin_amdgcn_wmma_f32_16x16x32_bf16(
                    false, a1.v, false, b.v, (short)0, acc[1][t], false, false);
            }
        }
        __syncthreads();
    }

    // C/D layout: VGPR i, lane l -> m = tileM + 8*(l>>4) + i, n = tileN + (l&15)
    const int mTop = mBase + ((lane >> 4) * 8);
    #pragma unroll
    for (int mf = 0; mf < 2; ++mf) {
        #pragma unroll
        for (int t = 0; t < 4; ++t) {
            int n = nBase + t * 16 + (lane & 15);
            if (n < N) {
                float bv = bias ? bias[n] : 0.0f;
                #pragma unroll
                for (int i = 0; i < 8; ++i)
                    C[(size_t)(mTop + mf * 16 + i) * N + n] = acc[mf][t][i] + bv;
            }
        }
    }
}

// ---------------------------------------------------------------------------
// dt = softplus(raw + dt_bias); dA = exp(dt * (-exp(A_log)))
// ---------------------------------------------------------------------------
__global__ void dt_kernel(const float* __restrict__ zxbcdt, const float* __restrict__ dt_bias,
                          const float* __restrict__ A_log,
                          float* __restrict__ dt, float* __restrict__ dA) {
    int idx = blockIdx.x * blockDim.x + threadIdx.x;   // NTOK * NHEADS
    if (idx >= NTOK * NHEADS) return;
    int h   = idx & (NHEADS - 1);
    int tok = idx >> 3;
    float x = zxbcdt[(size_t)tok * D_IN_PROJ + (D_INNER + CONV_DIM) + h] + dt_bias[h];
    float d = (x > 20.0f) ? x : log1pf(expf(x));
    dt[idx] = d;
    dA[idx] = expf(d * (-expf(A_log[h])));
}

// ---------------------------------------------------------------------------
// Causal depthwise conv (kernel 4) over xBC slice of zxbcdt, then SiLU
// ---------------------------------------------------------------------------
__global__ void conv_silu_kernel(const float* __restrict__ zxbcdt,
                                 const float* __restrict__ conv_w,
                                 const float* __restrict__ conv_b,
                                 float* __restrict__ xBC) {
    int idx = blockIdx.x * blockDim.x + threadIdx.x;   // NTOK * CONV_DIM
    if (idx >= NTOK * CONV_DIM) return;
    int c   = idx % CONV_DIM;
    int tok = idx / CONV_DIM;
    int l   = tok & (SEQLEN - 1);
    float acc = conv_b[c];
    #pragma unroll
    for (int k = 0; k < D_CONV; ++k) {
        int dl = l - (D_CONV - 1) + k;
        if (dl >= 0)
            acc += conv_w[k * CONV_DIM + c] *
                   zxbcdt[(size_t)(tok + k - (D_CONV - 1)) * D_IN_PROJ + D_INNER + c];
    }
    xBC[(size_t)tok * CONV_DIM + c] = acc / (1.0f + expf(-acc));
}

// ---------------------------------------------------------------------------
// Sequential SSM scan. One workgroup per (head, batch). 256 threads.
// Thread t owns state h[p][n] for p = t/4, n in [(t&3)*16, +16) -> 16 regs,
// kept in registers for all 2048 steps. Inputs staged in LDS 8 steps at a time.
// ---------------------------------------------------------------------------
__global__ void __launch_bounds__(256)
scan_kernel(const float* __restrict__ xBC, const float* __restrict__ dt,
            const float* __restrict__ dA, const float* __restrict__ Dp,
            float* __restrict__ y) {
    const int head = blockIdx.x;     // 0..NHEADS-1
    const int b    = blockIdx.y;     // 0..BATCH-1
    const int t    = threadIdx.x;    // 0..255
    const int p    = t >> 2;         // 0..63 (head dim)
    const int nb   = (t & 3) * 16;   // state-dim block

    float h[16];
    #pragma unroll
    for (int i = 0; i < 16; ++i) h[i] = 0.0f;

    __shared__ float sx[8][HEADDIM], sB[8][D_STATE], sC[8][D_STATE];
    __shared__ float sdt[8], sdA[8];
    const float Dh = Dp[head];

    for (int c0 = 0; c0 < SEQLEN; c0 += 8) {
        for (int j = t; j < 8 * 64; j += 256) {
            int s = j >> 6, e = j & 63;
            const float* row = xBC + (size_t)(b * SEQLEN + c0 + s) * CONV_DIM;
            sx[s][e] = row[head * HEADDIM + e];
            sB[s][e] = row[D_INNER + e];
            sC[s][e] = row[D_INNER + D_STATE + e];
        }
        if (t < 8) {
            int tok = b * SEQLEN + c0 + t;
            sdt[t] = dt[tok * NHEADS + head];
            sdA[t] = dA[tok * NHEADS + head];
        }
        __syncthreads();

        #pragma unroll
        for (int s = 0; s < 8; ++s) {
            const float dAv = sdA[s];
            const float xv  = sx[s][p];
            const float a   = sdt[s] * xv;
            float yp = 0.0f;
            #pragma unroll
            for (int i = 0; i < 16; ++i) {
                h[i] = h[i] * dAv + a * sB[s][nb + i];
                yp  += h[i] * sC[s][nb + i];
            }
            // reduce over the 4 lanes covering the state dimension (wave32)
            yp += __shfl_xor(yp, 1, 32);
            yp += __shfl_xor(yp, 2, 32);
            if ((t & 3) == 0) {
                int tok = b * SEQLEN + c0 + s;
                y[(size_t)tok * D_INNER + head * HEADDIM + p] = yp + Dh * xv;
            }
        }
        __syncthreads();
    }
}

// ---------------------------------------------------------------------------
// y = y * silu(z); gated RMSNorm over D_INNER; write bf16 for the next GEMM
// ---------------------------------------------------------------------------
__global__ void __launch_bounds__(256)
gatenorm_kernel(const float* __restrict__ y, const float* __restrict__ zxbcdt,
                const float* __restrict__ norm_w, __bf16* __restrict__ out) {
    const int tok = blockIdx.x;
    const int t   = threadIdx.x;
    float v[2];
    float ss = 0.0f;
    #pragma unroll
    for (int j = 0; j < 2; ++j) {
        int i  = t + j * 256;
        float z = zxbcdt[(size_t)tok * D_IN_PROJ + i];
        float val = y[(size_t)tok * D_INNER + i] * (z / (1.0f + expf(-z)));
        v[j] = val;
        ss  += val * val;
    }
    #pragma unroll
    for (int m = 1; m < 32; m <<= 1) ss += __shfl_xor(ss, m, 32);
    __shared__ float wsum[8];
    if ((t & 31) == 0) wsum[t >> 5] = ss;
    __syncthreads();
    float tot = 0.0f;
    #pragma unroll
    for (int w = 0; w < 8; ++w) tot += wsum[w];
    const float scale = rsqrtf(tot / (float)D_INNER + EPS);
    #pragma unroll
    for (int j = 0; j < 2; ++j) {
        int i = t + j * 256;
        out[(size_t)tok * D_INNER + i] = (__bf16)(v[j] * scale * norm_w[i]);
    }
}

// ---------------------------------------------------------------------------
// Host-side orchestration
// ---------------------------------------------------------------------------
namespace {

struct Ws {
    __bf16 *wInBf, *wOutBf, *wLinBf, *actBf;
    float  *zxbcdt, *xBCc, *dtB, *dAB, *yB, *m1, *h1;
};

static Ws carve(void* d_ws) {
    auto align256 = [](size_t x) { return (x + 255) & ~size_t(255); };
    char* base = (char*)d_ws;
    size_t off = 0;
    auto take = [&](size_t bytes) { void* p = base + off; off = align256(off + bytes); return p; };
    Ws w;
    w.wInBf  = (__bf16*)take((size_t)N_INPROJ_PAD * D_MODEL * 2);
    w.wOutBf = (__bf16*)take((size_t)D_MODEL * D_INNER * 2);
    w.wLinBf = (__bf16*)take((size_t)D_MODEL * D_MODEL * 2);
    w.actBf  = (__bf16*)take((size_t)NTOK * D_INNER * 2);
    w.zxbcdt = (float*)take((size_t)NTOK * D_IN_PROJ * 4);
    w.xBCc   = (float*)take((size_t)NTOK * CONV_DIM * 4);
    w.dtB    = (float*)take((size_t)NTOK * NHEADS * 4);
    w.dAB    = (float*)take((size_t)NTOK * NHEADS * 4);
    w.yB     = (float*)take((size_t)NTOK * D_INNER * 4);
    w.m1     = (float*)take((size_t)NTOK * D_MODEL * 4);
    w.h1     = (float*)take((size_t)NTOK * D_MODEL * 4);
    return w;
}

static void launch_gemm(const __bf16* A, const __bf16* W, float* C, const float* bias,
                        int M, int N, int K, hipStream_t s) {
    dim3 block(32, 8, 1);
    dim3 grid((N + 63) / 64, M / 256, 1);
    gemm_wmma_kernel<<<grid, block, 0, s>>>(A, W, C, bias, M, N, K);
}

static void mamba_pass(const float* in, float* out, const Ws& w,
                       const float* conv_w, const float* conv_b,
                       const float* dt_bias, const float* A_log,
                       const float* Dp, const float* norm_w,
                       hipStream_t s) {
    // 1) x -> bf16
    {
        int n = NTOK * D_MODEL;
        cvt_act_kernel<<<(n + 255) / 256, 256, 0, s>>>(in, w.actBf, n, 0);
    }
    // 2) zxbcdt = x @ in_proj_w^T   (M=8192, N=1160 (W padded to 1216), K=256)
    launch_gemm(w.actBf, w.wInBf, w.zxbcdt, nullptr, NTOK, D_IN_PROJ, D_MODEL, s);
    // 3) dt / dA
    {
        int n = NTOK * NHEADS;
        dt_kernel<<<(n + 255) / 256, 256, 0, s>>>(w.zxbcdt, dt_bias, A_log, w.dtB, w.dAB);
    }
    // 4) causal depthwise conv + silu
    {
        int n = NTOK * CONV_DIM;
        conv_silu_kernel<<<(n + 255) / 256, 256, 0, s>>>(w.zxbcdt, conv_w, conv_b, w.xBCc);
    }
    // 5) SSM scan
    scan_kernel<<<dim3(NHEADS, BATCH), 256, 0, s>>>(w.xBCc, w.dtB, w.dAB, Dp, w.yB);
    // 6) gate + RMSNorm -> bf16
    gatenorm_kernel<<<NTOK, 256, 0, s>>>(w.yB, w.zxbcdt, norm_w, w.actBf);
    // 7) out = yn @ out_proj_w^T   (M=8192, N=256, K=512)
    launch_gemm(w.actBf, w.wOutBf, out, nullptr, NTOK, D_MODEL, D_INNER, s);
}

} // namespace

extern "C" void kernel_launch(void* const* d_in, const int* in_sizes, int n_in,
                              void* d_out, int out_size, void* d_ws, size_t ws_size,
                              hipStream_t stream) {
    (void)in_sizes; (void)n_in; (void)out_size; (void)ws_size;

    const float* x         = (const float*)d_in[0];
    const float* in_proj_w = (const float*)d_in[1];
    const float* conv_w    = (const float*)d_in[2];
    const float* conv_b    = (const float*)d_in[3];
    const float* dt_bias   = (const float*)d_in[4];
    const float* A_log     = (const float*)d_in[5];
    const float* Dp        = (const float*)d_in[6];
    const float* norm_w    = (const float*)d_in[7];
    const float* out_proj_w= (const float*)d_in[8];
    const float* linear_w  = (const float*)d_in[9];
    const float* linear_b  = (const float*)d_in[10];
    float* out = (float*)d_out;

    Ws w = carve(d_ws);

    // Weight conversions (every launch: no cached state allowed)
    {
        int n = N_INPROJ_PAD * D_MODEL;
        cvt_weight_kernel<<<(n + 255) / 256, 256, 0, stream>>>(in_proj_w, w.wInBf,
                                                               D_IN_PROJ, D_MODEL, N_INPROJ_PAD);
        n = D_MODEL * D_INNER;
        cvt_weight_kernel<<<(n + 255) / 256, 256, 0, stream>>>(out_proj_w, w.wOutBf,
                                                               D_MODEL, D_INNER, D_MODEL);
        n = D_MODEL * D_MODEL;
        cvt_weight_kernel<<<(n + 255) / 256, 256, 0, stream>>>(linear_w, w.wLinBf,
                                                               D_MODEL, D_MODEL, D_MODEL);
    }

    // ---- pass 1: mamba2(x) -> m1
    mamba_pass(x, w.m1, w, conv_w, conv_b, dt_bias, A_log, Dp, norm_w, stream);

    // ---- h1 = leaky_relu(m1) @ linear_w^T + linear_b
    {
        int n = NTOK * D_MODEL;
        cvt_act_kernel<<<(n + 255) / 256, 256, 0, stream>>>(w.m1, w.actBf, n, /*leaky=*/1);
        launch_gemm(w.actBf, w.wLinBf, w.h1, linear_b, NTOK, D_MODEL, D_MODEL, stream);
    }

    // ---- pass 2: mamba2(h1) -> m1 (reuse)
    mamba_pass(w.h1, w.m1, w, conv_w, conv_b, dt_bias, A_log, Dp, norm_w, stream);

    // ---- out = m1 @ linear_w^T + linear_b
    {
        int n = NTOK * D_MODEL;
        cvt_act_kernel<<<(n + 255) / 256, 256, 0, stream>>>(w.m1, w.actBf, n, /*leaky=*/0);
        launch_gemm(w.actBf, w.wLinBf, out, linear_b, NTOK, D_MODEL, D_MODEL, stream);
    }
}
